// Zero_Parameter_87299505258559
// MI455X (gfx1250) — matrix operand order repaired and checked by reading, not compiled, not run on hardware
//
#include <hip/hip_runtime.h>
#include <hip/hip_bf16.h>

typedef _Float16 v16h __attribute__((ext_vector_type(16)));
typedef _Float16 v8h  __attribute__((ext_vector_type(8)));
typedef float    v8f  __attribute__((ext_vector_type(8)));
typedef int      b128_t __attribute__((vector_size(16)));   // matches builtin param

#define CDIM  1024
#define N1T   1024
#define N2T   4096
#define HD    128
#define SCALEF 0.03125f   // 1024^-0.5

// ---- CDNA5 async global->LDS copy (ASYNCcnt), with safe fallback ---------
#if defined(__HIP_DEVICE_COMPILE__) && __has_builtin(__builtin_amdgcn_global_load_async_to_lds_b128)
#define USE_ASYNC_LDS 1
#else
#define USE_ASYNC_LDS 0
#endif

__device__ __forceinline__ void cp16_g2l(void* lds, const void* g) {
#if USE_ASYNC_LDS
  __builtin_amdgcn_global_load_async_to_lds_b128(
      (__attribute__((address_space(1))) b128_t*)(uintptr_t)g,
      (__attribute__((address_space(3))) b128_t*)(uint32_t)(uintptr_t)lds,
      0, 0);
#else
  *reinterpret_cast<v8h*>(lds) = *reinterpret_cast<const v8h*>(g);
#endif
}

__device__ __forceinline__ void wait_async_all() {
#if USE_ASYNC_LDS
#if __has_builtin(__builtin_amdgcn_s_wait_asynccnt)
  __builtin_amdgcn_s_wait_asynccnt(0);
#else
  asm volatile("s_wait_asynccnt 0x0" ::: "memory");
#endif
#endif
}

__device__ __forceinline__ v8f wmma16(v16h a, v16h b, v8f c) {
  return __builtin_amdgcn_wmma_f32_16x16x32_f16(false, a, false, b, (short)0, c,
                                                false, false);
}

// ---- WMMA fragment loaders (layouts per CDNA5 ISA 7.12.2) ----------------
__device__ __forceinline__ v16h frag_a_h(const _Float16* __restrict__ base,
                                         int ld, int r0, int k0, int lane) {
  const int m = lane & 15, g = lane >> 4;
  const _Float16* p = base + (size_t)(r0 + m) * ld + k0 + g * 8;
  v8h lo = *reinterpret_cast<const v8h*>(p);
  v8h hi = *reinterpret_cast<const v8h*>(p + 16);
  v16h a;
#pragma unroll
  for (int i = 0; i < 8; ++i) { a[i] = lo[i]; a[i + 8] = hi[i]; }
  return a;
}

__device__ __forceinline__ v16h frag_b_lds(const _Float16* p) {
  v8h lo = *reinterpret_cast<const v8h*>(p);
  v8h hi = *reinterpret_cast<const v8h*>(p + 16);
  v16h b;
#pragma unroll
  for (int i = 0; i < 8; ++i) { b[i] = lo[i]; b[i + 8] = hi[i]; }
  return b;
}

__device__ __forceinline__ v16h frag_a_f(const float* __restrict__ base,
                                         int ld, int r0, int k0, int lane) {
  const int m = lane & 15, g = lane >> 4;
  const float* p = base + (size_t)(r0 + m) * ld + k0 + g * 8;
  v16h a;
#pragma unroll
  for (int i = 0; i < 8; ++i) { a[i] = (_Float16)p[i]; a[i + 8] = (_Float16)p[16 + i]; }
  return a;
}

__device__ __forceinline__ v16h frag_b_f(const float* __restrict__ base,
                                         int ld, int n0, int k0, int lane) {
  const int n = lane & 15, g = lane >> 4;
  const float* p = base + (size_t)(n0 + n) * ld + k0 + g * 8;
  v16h b;
#pragma unroll
  for (int i = 0; i < 16; ++i) b[i] = (_Float16)p[i < 8 ? i : i + 8];
  return b;
}

// ---- Projection GEMM: Y = X * W^T + bias  --------------------------------
// Block = 8 waves, 128x128 tile; wave tile 32x64 (2x4 WMMA accumulators).
// MODE 0: Y row-major f16 [row][col]   (Q, K)
// MODE 1: Y per-head transposed f16: Y[(b*1024+col)*4096 + token]  (V^T)
template <int MODE>
__global__ __launch_bounds__(256) void proj_kernel(
    const float* __restrict__ X, const float* __restrict__ W,
    const float* __restrict__ bias, _Float16* __restrict__ Y) {
  const int lane = threadIdx.x & 31;
  const int wave = threadIdx.x >> 5;
  const int rowbase = blockIdx.x * 128 + (wave & 3) * 32;
  const int colbase = blockIdx.y * 128 + (wave >> 2) * 64;

  v8f acc[2][4];
#pragma unroll
  for (int i = 0; i < 2; ++i)
#pragma unroll
    for (int j = 0; j < 4; ++j) acc[i][j] = (v8f){0, 0, 0, 0, 0, 0, 0, 0};

  for (int c0 = 0; c0 < CDIM; c0 += 32) {
    v16h a0 = frag_a_f(X, CDIM, rowbase, c0, lane);
    v16h a1 = frag_a_f(X, CDIM, rowbase + 16, c0, lane);
    v16h bfr[4];
#pragma unroll
    for (int nt = 0; nt < 4; ++nt) bfr[nt] = frag_b_f(W, CDIM, colbase + nt * 16, c0, lane);
#pragma unroll
    for (int nt = 0; nt < 4; ++nt) {
      acc[0][nt] = wmma16(a0, bfr[nt], acc[0][nt]);
      acc[1][nt] = wmma16(a1, bfr[nt], acc[1][nt]);
    }
  }

  const int n = lane & 15, g = lane >> 4;
#pragma unroll
  for (int mt = 0; mt < 2; ++mt) {
#pragma unroll
    for (int nt = 0; nt < 4; ++nt) {
      const int col = colbase + nt * 16 + n;
      const float bv = bias[col];
      const int row0 = rowbase + mt * 16 + g * 8;
      if (MODE == 0) {
#pragma unroll
        for (int r = 0; r < 8; ++r)
          Y[(size_t)(row0 + r) * CDIM + col] = (_Float16)(acc[mt][nt][r] + bv);
      } else {
        v8h pk;
#pragma unroll
        for (int r = 0; r < 8; ++r) pk[r] = (_Float16)(acc[mt][nt][r] + bv);
        const int bidx = row0 >> 12;      // token row / 4096 = batch
        const int mtok = row0 & 4095;     // token within batch (8-aligned)
        *reinterpret_cast<v8h*>(Y + ((size_t)(bidx * CDIM + col)) * N2T + mtok) = pk;
      }
    }
  }
}

// ---- Flash attention: 4 waves/block, 64 q rows, async double-buffered K/V
__global__ __launch_bounds__(128) void attn_kernel(
    const _Float16* __restrict__ Qf, const _Float16* __restrict__ Kf,
    const _Float16* __restrict__ Vt, const float* __restrict__ F_t,
    float* __restrict__ out) {
  const int tid  = threadIdx.x;
  const int lane = tid & 31;
  const int wv   = tid >> 5;          // wave id 0..3 -> q-row subtile
  const int blk  = blockIdx.x;        // 256 blocks
  const int qblk = blk & 15;
  const int h    = (blk >> 4) & 7;
  const int b    = blk >> 7;
  const int g    = lane >> 4, nn = lane & 15;

  __shared__ _Float16 Kt[2][64][128];     // [buf][key][c]        2x16 KB
  __shared__ _Float16 Vtile[2][128][64];  // [buf][c][key]        2x16 KB
  __shared__ float    Sbuf[4][16][64];    // per-wave scores        16 KB
  __shared__ _Float16 Pbuf[4][16][64];    // per-wave probs (f16)    8 KB

  const _Float16* Qbase = Qf + ((size_t)(b * N1T + qblk * 64 + wv * 16)) * CDIM + h * HD;
  const _Float16* Kbase = Kf + (size_t)b * N2T * CDIM + h * HD;
  const _Float16* Vbase = Vt + ((size_t)(b * CDIM + h * HD)) * N2T;

  // async cooperative stage of one 64-key chunk (K: 64x128, V^T: 128x64)
  auto stage = [&](int bf, int m0s) {
#pragma unroll
    for (int rep = 0; rep < 8; ++rep) {
      const int idx = rep * 128 + tid;          // 0..1023, 16B segments
      const int kr = idx >> 4, ks = idx & 15;
      cp16_g2l(&Kt[bf][kr][ks * 8], Kbase + (size_t)(m0s + kr) * CDIM + ks * 8);
    }
#pragma unroll
    for (int rep = 0; rep < 8; ++rep) {
      const int idx = rep * 128 + tid;
      const int vr = idx >> 3, vs = idx & 7;
      cp16_g2l(&Vtile[bf][vr][vs * 8], Vbase + (size_t)vr * N2T + m0s + vs * 8);
    }
  };

  v16h qfrag[4];
#pragma unroll
  for (int ct = 0; ct < 4; ++ct) qfrag[ct] = frag_a_h(Qbase, CDIM, 0, ct * 32, lane);

  v8f oacc[8];
#pragma unroll
  for (int t = 0; t < 8; ++t) oacc[t] = (v8f){0, 0, 0, 0, 0, 0, 0, 0};

  float mrow = -3.0e38f, lrow = 0.f;

  stage(0, 0);                       // prefetch chunk 0
  int buf = 0;
  for (int m0 = 0; m0 < N2T; m0 += 64) {
    wait_async_all();
    __syncthreads();                 // staged chunk visible; prev chunk consumed
    if (m0 + 64 < N2T) stage(buf ^ 1, m0 + 64);   // overlap next chunk's DMA

    // ---- S = Q K^T * scale for this wave's 16 q-rows x 64 keys ----------
#pragma unroll
    for (int kt = 0; kt < 4; ++kt) {
      v16h kb[4];
#pragma unroll
      for (int ct = 0; ct < 4; ++ct)
        kb[ct] = frag_b_lds(&Kt[buf][kt * 16 + nn][ct * 32 + g * 8]);
      v8f s = (v8f){0, 0, 0, 0, 0, 0, 0, 0};
#pragma unroll
      for (int ct = 0; ct < 4; ++ct) s = wmma16(qfrag[ct], kb[ct], s);
#pragma unroll
      for (int r = 0; r < 8; ++r) Sbuf[wv][g * 8 + r][kt * 16 + nn] = s[r] * SCALEF;
    }
    __syncthreads();

    // ---- online softmax: lane owns row nn, columns [g*32, g*32+32) ------
    float cmax = -3.0e38f;
#pragma unroll 8
    for (int j = 0; j < 32; ++j) cmax = fmaxf(cmax, Sbuf[wv][nn][g * 32 + j]);
    cmax = fmaxf(cmax, __shfl_xor(cmax, 16, 32));
    const float mnew  = fmaxf(mrow, cmax);
    const float alpha = __expf(mrow - mnew);
    float csum = 0.f;
#pragma unroll 8
    for (int j = 0; j < 32; ++j) {
      float p = __expf(Sbuf[wv][nn][g * 32 + j] - mnew);
      csum += p;
      Pbuf[wv][nn][g * 32 + j] = (_Float16)p;
    }
    csum += __shfl_xor(csum, 16, 32);
    lrow = lrow * alpha + csum;
    mrow = mnew;
    __syncthreads();

    // ---- rescale O: row M = r + g*8; alpha for row M lives in lane M ----
    float ar[8];
#pragma unroll
    for (int r = 0; r < 8; ++r) ar[r] = __shfl(alpha, g * 8 + r, 32);
#pragma unroll
    for (int t = 0; t < 8; ++t)
#pragma unroll
      for (int r = 0; r < 8; ++r) oacc[t][r] *= ar[r];

    // ---- O += P V  (P from LDS in A layout, V^T tile gives B frags) -----
#pragma unroll
    for (int kk = 0; kk < 2; ++kk) {
      const _Float16* pp = &Pbuf[wv][nn][kk * 32 + g * 8];
      v8h lo = *reinterpret_cast<const v8h*>(pp);
      v8h hi = *reinterpret_cast<const v8h*>(pp + 16);
      v16h pa;
#pragma unroll
      for (int i = 0; i < 8; ++i) { pa[i] = lo[i]; pa[i + 8] = hi[i]; }
#pragma unroll
      for (int th = 0; th < 2; ++th) {
        v16h vb[4];
#pragma unroll
        for (int j = 0; j < 4; ++j)
          vb[j] = frag_b_lds(&Vtile[buf][(th * 4 + j) * 16 + nn][kk * 32 + g * 8]);
#pragma unroll
        for (int j = 0; j < 4; ++j)
          oacc[th * 4 + j] = wmma16(pa, vb[j], oacc[th * 4 + j]);
      }
    }
    buf ^= 1;
  }

  // ---- epilogue: divide by row sum, add residual -------------------------
  const float linv = 1.0f / lrow;
  float lr[8];
#pragma unroll
  for (int r = 0; r < 8; ++r) lr[r] = __shfl(linv, g * 8 + r, 32);
#pragma unroll
  for (int t = 0; t < 8; ++t) {
#pragma unroll
    for (int r = 0; r < 8; ++r) {
      const int grow = b * N1T + qblk * 64 + wv * 16 + g * 8 + r;
      const int col  = h * HD + t * 16 + nn;
      const size_t idx = (size_t)grow * CDIM + col;
      out[idx] = oacc[t][r] * lr[r] + F_t[idx];
    }
  }
}

extern "C" void kernel_launch(void* const* d_in, const int* in_sizes, int n_in,
                              void* d_out, int out_size, void* d_ws, size_t ws_size,
                              hipStream_t stream) {
  (void)in_sizes; (void)n_in; (void)out_size; (void)ws_size;
  const float* F_t = (const float*)d_in[0];
  const float* F_s = (const float*)d_in[1];
  const float* Wq  = (const float*)d_in[2];
  const float* bq  = (const float*)d_in[3];
  const float* Wk  = (const float*)d_in[4];
  const float* bk  = (const float*)d_in[5];
  const float* Wv  = (const float*)d_in[6];
  const float* bv  = (const float*)d_in[7];
  float* out = (float*)d_out;

  char* ws = (char*)d_ws;
  const size_t QBYTES = (size_t)2 * N1T * CDIM * sizeof(_Float16);  // 4 MB
  const size_t KBYTES = (size_t)2 * N2T * CDIM * sizeof(_Float16);  // 16 MB
  _Float16* Qf = (_Float16*)ws;
  _Float16* Kf = (_Float16*)(ws + QBYTES);
  _Float16* Vt = (_Float16*)(ws + QBYTES + KBYTES);

  // Q: 2048 rows, K: 8192 rows, V: 8192 rows (transposed per-head output)
  proj_kernel<0><<<dim3(2048 / 128, CDIM / 128), 256, 0, stream>>>(F_t, Wq, bq, Qf);
  proj_kernel<0><<<dim3(8192 / 128, CDIM / 128), 256, 0, stream>>>(F_s, Wk, bk, Kf);
  proj_kernel<1><<<dim3(8192 / 128, CDIM / 128), 256, 0, stream>>>(F_s, Wv, bv, Vt);

  // 2 batches * 8 heads * 16 q-blocks of 64 rows = 256 blocks, 4 waves each
  attn_kernel<<<dim3(256), dim3(128), 0, stream>>>(Qf, Kf, Vt, F_t, out);
}
